// SingleRenderer_78709570667300
// MI455X (gfx1250) — matrix operand order repaired and checked by reading, not compiled, not run on hardware
//
#include <hip/hip_runtime.h>
#include <hip/hip_bf16.h>
#include <math.h>

#define NRAYS 2048
#define SC 64
#define SF 128
#define HID 256
#define USE_ASYNC_LDS 1

typedef _Float16 v16h __attribute__((ext_vector_type(16)));
typedef float    v8f  __attribute__((ext_vector_type(8)));

struct MlpW { const float *W1,*b1,*b2,*Wsig,*bsig,*Wrgb,*brgb; };

__device__ __forceinline__ float softplusf_(float x){ return (x > 20.0f) ? x : log1pf(expf(x)); }
__device__ __forceinline__ float sigmoidf_(float x){ return 1.0f/(1.0f+expf(-x)); }

// ---------------------------------------------------------------------------
// One-shot: convert W2 [256x256] f32 row-major -> f16 transposed [col][k]
// ---------------------------------------------------------------------------
__global__ void k_w2cvt(const float* __restrict__ src, _Float16* __restrict__ dst){
  int idx = blockIdx.x*blockDim.x + threadIdx.x;      // 65536 threads
  int n = idx >> 8, k = idx & 255;
  dst[idx] = (_Float16)src[k*HID + n];                // dst[n*256 + k] = W2[k][n]
}

// ---------------------------------------------------------------------------
// Ray setup: normalize dirs, sphere near/far, coarse fg/bg sample depths
// ---------------------------------------------------------------------------
__global__ void k_ray_setup(const float* __restrict__ ro, const float* __restrict__ rd,
                            float* __restrict__ dn, float* __restrict__ far,
                            float* __restrict__ dfg, float* __restrict__ dbg){
  int i = blockIdx.x*blockDim.x + threadIdx.x;
  if (i >= NRAYS) return;
  float ox=ro[3*i], oy=ro[3*i+1], oz=ro[3*i+2];
  float dx=rd[3*i], dy=rd[3*i+1], dz=rd[3*i+2];
  float inv = rsqrtf(dx*dx+dy*dy+dz*dz);
  dx*=inv; dy*=inv; dz*=inv;
  dn[3*i]=dx; dn[3*i+1]=dy; dn[3*i+2]=dz;
  float dop = ox*dx+oy*dy+oz*dz;
  float oo  = ox*ox+oy*oy+oz*oz;
  float under = dop*dop - (oo - 1.0f);
  float sq = sqrtf(fmaxf(under, 1e-10f));
  float nr = fmaxf(-dop - sq, 1e-3f);
  float fr = -dop + sq;
  far[i] = fr;
  for (int s=0;s<SC;++s){
    float t = (float)s * (1.0f/63.0f);
    dfg[i*SC+s] = nr*(1.0f-t) + fr*t;
    float tb = (float)(SC - s) * (1.0f/65.0f);   // flipped linspace(0,1,66)[1:-1]
    dbg[i*SC+s] = fr / tb;
  }
}

// ---------------------------------------------------------------------------
// Fused MLP: 64 points/block, 8 wave32 waves. Layer2 = WMMA f16 (f32 accum),
// W2 staged per 32-wide k-slice via async global->LDS (pre-converted f16^T).
// ---------------------------------------------------------------------------
__global__ __launch_bounds__(256)
void k_mlp(const float* __restrict__ ro, const float* __restrict__ dn,
           const float* __restrict__ depth, int sShift, int isBg, MlpW w,
           const _Float16* __restrict__ W2T,
           float* __restrict__ sig_out, float* __restrict__ rgb_out){
  __shared__ float    xin[64][4];
  __shared__ float    vin[64][3];
  __shared__ _Float16 hs [64][264];   // h1 tile, reused for h2 (stride 264: 16B-aligned rows, conflict-free)
  __shared__ _Float16 w2s[HID][40];   // k-slice transposed [col][kk], stride 40 (80B: aligned + conflict-free)

  const int tid = threadIdx.x;
  const int p0  = blockIdx.x * 64;
  const int din = 3 + isBg;

  if (tid < 64){
    int p = p0 + tid;
    int r = p >> sShift;
    float t  = depth[p];
    float dx = dn[3*r], dy = dn[3*r+1], dz = dn[3*r+2];
    float px = ro[3*r] + dx*t, py = ro[3*r+1] + dy*t, pz = ro[3*r+2] + dz*t;
    if (isBg){
      float rr = sqrtf(px*px+py*py+pz*pz);
      float ir = 1.0f/rr;
      xin[tid][0]=px*ir; xin[tid][1]=py*ir; xin[tid][2]=pz*ir; xin[tid][3]=ir;
    } else {
      xin[tid][0]=px; xin[tid][1]=py; xin[tid][2]=pz; xin[tid][3]=0.0f;
    }
    vin[tid][0]=dx; vin[tid][1]=dy; vin[tid][2]=dz;
  }
  __syncthreads();

  // ----- layer 1 (din <= 4, VALU) -> f16 LDS tile -----
  for (int idx = tid; idx < 64*HID; idx += 256){
    int p = idx >> 8, j = idx & 255;
    float a = w.b1[j];
    for (int k=0;k<din;++k) a += xin[p][k]*w.W1[k*HID+j];
    hs[p][j] = (_Float16)fmaxf(a, 0.0f);
  }
  __syncthreads();

  // ----- layer 2: [64 x 256] @ [256 x 256] via v_wmma_f32_16x16x32_f16 -----
  const int lane = tid & 31, wv = tid >> 5;
  const int hi = lane >> 4, l15 = lane & 15;
  const int m0 = (wv & 3) * 16;        // 4 row groups
  const int c0 = (wv >> 2) * 128;      // 2 col groups of 128
  const int m  = m0 + l15;

  v8f acc[8];
  for (int ct=0; ct<8; ++ct)
    for (int e=0; e<8; ++e) acc[ct][e] = 0.0f;

  const unsigned ldsoff = (unsigned)(uintptr_t)(&w2s[tid][0]);

  for (int k0 = 0; k0 < HID; k0 += 32){
    const _Float16* gsl = W2T + (size_t)tid*HID + k0;   // this thread's column slice (64B)
#if USE_ASYNC_LDS
    asm volatile(
      "global_load_async_to_lds_b128 %0, %1, off\n\t"
      "global_load_async_to_lds_b128 %0, %1, off offset:16\n\t"
      "global_load_async_to_lds_b128 %0, %1, off offset:32\n\t"
      "global_load_async_to_lds_b128 %0, %1, off offset:48"
      :: "v"(ldsoff), "v"(gsl) : "memory");
    if (k0 + 32 < HID) __builtin_prefetch(gsl + 32, 0, 1);
    asm volatile("s_wait_asynccnt 0" ::: "memory");
#else
    {
      const uint4* g4 = (const uint4*)gsl;
      uint4* l4 = (uint4*)&w2s[tid][0];
      l4[0]=g4[0]; l4[1]=g4[1]; l4[2]=g4[2]; l4[3]=g4[3];
      if (k0 + 32 < HID) __builtin_prefetch(gsl + 32, 0, 1);
    }
#endif
    __syncthreads();

    // A fragment (16x32 f16): two aligned 16B chunks per lane
    union { v16h v; uint4 q[2]; } au;
    au.q[0] = *(const uint4*)&hs[m][k0 + 8*hi];
    au.q[1] = *(const uint4*)&hs[m][k0 + 8*hi + 16];
#pragma unroll
    for (int ct=0; ct<8; ++ct){
      int col = c0 + ct*16 + l15;
      union { v16h v; uint4 q[2]; } bu;    // B 32x16: lane holds K 8h..8h+7 and 16+8h..16+8h+7 of its column
      bu.q[0] = *(const uint4*)&w2s[col][8*hi];
      bu.q[1] = *(const uint4*)&w2s[col][8*hi + 16];
      acc[ct] = __builtin_amdgcn_wmma_f32_16x16x32_f16(
          false, au.v, false, bu.v, (short)0, acc[ct], false, false);
    }
    __syncthreads();
  }

  // bias + relu, write h2 over the h1 tile (all h1 reads completed at barrier)
#pragma unroll
  for (int ct=0; ct<8; ++ct){
    int n = c0 + ct*16 + l15;
#pragma unroll
    for (int r=0;r<8;++r){
      int mm = m0 + r + 8*hi;
      float v = acc[ct][r] + w.b2[n];
      hs[mm][n] = (_Float16)fmaxf(v, 0.0f);
    }
  }
  __syncthreads();

  // ----- heads: sigma and sigmoid rgb (Wrgb rows 256..258 = view dirs) -----
  {
    int p = tid >> 2, o = tid & 3;
    if (o == 0){
      float a = w.bsig[0];
      for (int j=0;j<HID;++j) a += (float)hs[p][j] * w.Wsig[j];
      sig_out[p0+p] = a;                       // raw; softplus in composite
    } else {
      int c = o - 1;
      float a = w.brgb[c];
      for (int j=0;j<HID;++j) a += (float)hs[p][j] * w.Wrgb[j*3+c];
      for (int v2=0; v2<3; ++v2) a += vin[p][v2] * w.Wrgb[(HID+v2)*3+c];
      rgb_out[(p0+p)*3+c] = sigmoidf_(a);
    }
  }
}

// ---------------------------------------------------------------------------
// Alpha compositing (fg + bg_lambda * bg), autoexpo scale/shift
// ---------------------------------------------------------------------------
__global__ void k_composite(const float* __restrict__ sf, const float* __restrict__ rf,
                            const float* __restrict__ sb, const float* __restrict__ rb,
                            const float* __restrict__ dfg, const float* __restrict__ dbg,
                            const float* __restrict__ far,
                            const float* __restrict__ scale, const float* __restrict__ shift,
                            int S,
                            float* __restrict__ out_rgb, float* __restrict__ out_depth,
                            float* __restrict__ out_acc,
                            float* __restrict__ wfg, float* __restrict__ wbg){
  int i = blockIdx.x*blockDim.x + threadIdx.x;
  if (i >= NRAYS) return;
  float T = 1.0f, r0=0,r1=0,r2=0, dep=0, ac=0;
  for (int s=0;s<S;++s){
    float d = dfg[i*S+s];
    float dist = (s < S-1) ? (dfg[i*S+s+1] - d) : (far[i] - d);
    float alpha = 1.0f - expf(-softplusf_(sf[i*S+s]) * dist);
    float wgt = alpha * T;
    r0 += wgt*rf[(i*S+s)*3]; r1 += wgt*rf[(i*S+s)*3+1]; r2 += wgt*rf[(i*S+s)*3+2];
    dep += wgt * d; ac += wgt;
    T *= (1.0f - alpha + 1e-10f);
    if (wfg) wfg[i*S+s] = wgt;
  }
  float lam = T;                       // bg_lambda = T[..., -1]
  float Tb = 1.0f, b0=0,b1=0,b2=0, depb=0, acb=0;
  for (int s=0;s<S;++s){
    float d = dbg[i*S+s];
    float dist = (s < S-1) ? (dbg[i*S+s+1] - d) : 1e10f;
    float alpha = 1.0f - expf(-softplusf_(sb[i*S+s]) * dist);
    float wgt = alpha * Tb;
    b0 += wgt*rb[(i*S+s)*3]; b1 += wgt*rb[(i*S+s)*3+1]; b2 += wgt*rb[(i*S+s)*3+2];
    depb += wgt*d; acb += wgt;
    Tb *= (1.0f - alpha + 1e-10f);
    if (wbg) wbg[i*S+s] = wgt;
  }
  float sc = scale[0], sh = shift[0];
  out_rgb[3*i]   = (r0 + lam*b0 - sh)/sc;
  out_rgb[3*i+1] = (r1 + lam*b1 - sh)/sc;
  out_rgb[3*i+2] = (r2 + lam*b2 - sh)/sc;
  out_depth[i] = dep + lam*depb;
  out_acc[i]   = ac  + lam*acb;
}

// ---------------------------------------------------------------------------
// Hierarchical inverse-CDF sampling + merge-sort with coarse depths
// ---------------------------------------------------------------------------
__device__ __forceinline__ float rnd01(unsigned a, unsigned b, unsigned c){
  unsigned x = a*0x9E3779B1u ^ (b + 0x7F4A7C15u)*0x85EBCA77u ^ c*0xC2B2AE3Du;
  x ^= x>>16; x *= 0x7FEB352Du; x ^= x>>15; x *= 0x846CA68Bu; x ^= x>>16;
  return (float)(x >> 8) * (1.0f/16777216.0f);
}

__global__ void k_sample_pdf(const float* __restrict__ din_, const float* __restrict__ w,
                             float* __restrict__ dout, unsigned salt){
  int i = blockIdx.x*blockDim.x + threadIdx.x;
  if (i >= NRAYS) return;
  float cdf[63], bins[63], smp[SC];
  for (int j=0;j<63;++j) bins[j] = 0.5f*(din_[i*SC+j] + din_[i*SC+j+1]);
  cdf[0] = 0.0f;
  float tot = 0.0f;
  for (int j=0;j<62;++j){ tot += w[i*SC + j + 1] + 1e-5f; cdf[j+1] = tot; }
  float invt = 1.0f/tot;
  for (int j=1;j<63;++j) cdf[j] *= invt;
  for (int k=0;k<SC;++k){
    float u = rnd01(salt, (unsigned)i, (unsigned)k);
    int ind = 0;
    for (int j=0;j<63;++j) if (cdf[j] <= u) ind = j+1;   // searchsorted right
    int below = min(max(ind-1,0), 62);
    int above = min(ind, 62);
    float den = cdf[above]-cdf[below];
    if (den < 1e-5f) den = 1.0f;
    float t = (u - cdf[below])/den;
    smp[k] = bins[below] + t*(bins[above]-bins[below]);
  }
  for (int k=1;k<SC;++k){                 // insertion sort samples
    float v = smp[k]; int j=k-1;
    while (j>=0 && smp[j] > v){ smp[j+1]=smp[j]; --j; }
    smp[j+1]=v;
  }
  int a=0,b=0;                            // merge (coarse depths already sorted)
  for (int k=0;k<2*SC;++k){
    float va = (a<SC)? din_[i*SC+a] : 3.4e38f;
    float vb = (b<SC)? smp[b]       : 3.4e38f;
    if (va <= vb){ dout[i*2*SC+k]=va; ++a; } else { dout[i*2*SC+k]=vb; ++b; }
  }
}

// ---------------------------------------------------------------------------
extern "C" void kernel_launch(void* const* d_in, const int* in_sizes, int n_in,
                              void* d_out, int out_size, void* d_ws, size_t ws_size,
                              hipStream_t stream){
  (void)n_in; (void)out_size; (void)ws_size;
  auto f = [&](int i)->const float*{ return (const float*)d_in[i]; };

  // Detect flatten convention: insertion order (rays first) vs sorted keys
  int iRo, iRd, base;
  if (in_sizes[0] == NRAYS*3){ iRo = 0; iRd = 1; base = 2; }
  else { base = 0; iRd = 36; iRo = 37; }            // sorted: params < rays_d < rays_o
  bool sortedLeaves = (in_sizes[base+1] == HID*HID); // sorted leaf order: W1,W2,Wrgb,Wsig,b1,b2,brgb,bsig
  auto mlp = [&](int b, const float** W2)->MlpW{
    MlpW m;
    if (!sortedLeaves){ m.W1=f(b);   m.b1=f(b+1); *W2=f(b+2);   m.b2=f(b+3);
                        m.Wsig=f(b+4); m.bsig=f(b+5); m.Wrgb=f(b+6); m.brgb=f(b+7); }
    else              { m.W1=f(b);   *W2=f(b+1); m.Wrgb=f(b+2); m.Wsig=f(b+3);
                        m.b1=f(b+4); m.b2=f(b+5); m.brgb=f(b+6); m.bsig=f(b+7); }
    return m;
  };
  bool fgFirst = (in_sizes[base] == 3*HID);          // fg.W1 = 768, bg.W1 = 1024
  int cFg = fgFirst ? base      : base+8;
  int cBg = fgFirst ? base+8    : base;
  int cScale = base+16, cShift = base+17;
  int fFg = fgFirst ? base+18   : base+26;
  int fBg = fgFirst ? base+26   : base+18;
  int fScale = base+34, fShift = base+35;

  // workspace carve (floats)
  float* W    = (float*)d_ws;
  float* dn   = W; W += NRAYS*3;
  float* far  = W; W += NRAYS;
  float* dfgc = W; W += NRAYS*SC;
  float* dbgc = W; W += NRAYS*SC;
  float* dfgf = W; W += NRAYS*SF;
  float* dbgf = W; W += NRAYS*SF;
  float* sfg  = W; W += NRAYS*SF;
  float* sbg  = W; W += NRAYS*SF;
  float* rfg  = W; W += NRAYS*SF*3;
  float* rbg  = W; W += NRAYS*SF*3;
  float* wfg  = W; W += NRAYS*SC;
  float* wbg  = W; W += NRAYS*SC;
  float* tmp  = W; W += NRAYS*5;                     // coarse rgb/depth/acc (unused)
  _Float16* w2t[4];
  for (int q=0;q<4;++q){ w2t[q] = (_Float16*)W; W += HID*HID/2; }

  const float *W2cf, *W2cb, *W2ff, *W2fb;
  MlpW mCf = mlp(cFg, &W2cf), mCb = mlp(cBg, &W2cb);
  MlpW mFf = mlp(fFg, &W2ff), mFb = mlp(fBg, &W2fb);

  dim3 b256(256);
  dim3 gRay((NRAYS+255)/256);
  dim3 gCvt(HID*HID/256);

  // one-shot weight conversion (f32 row-major -> f16 transposed)
  k_w2cvt<<<gCvt, b256, 0, stream>>>(W2cf, w2t[0]);
  k_w2cvt<<<gCvt, b256, 0, stream>>>(W2cb, w2t[1]);
  k_w2cvt<<<gCvt, b256, 0, stream>>>(W2ff, w2t[2]);
  k_w2cvt<<<gCvt, b256, 0, stream>>>(W2fb, w2t[3]);

  k_ray_setup<<<gRay, b256, 0, stream>>>(f(iRo), f(iRd), dn, far, dfgc, dbgc);

  // coarse (sShift = log2(64) = 6)
  k_mlp<<<dim3(NRAYS*SC/64), b256, 0, stream>>>(f(iRo), dn, dfgc, 6, 0, mCf, w2t[0], sfg, rfg);
  k_mlp<<<dim3(NRAYS*SC/64), b256, 0, stream>>>(f(iRo), dn, dbgc, 6, 1, mCb, w2t[1], sbg, rbg);
  k_composite<<<gRay, b256, 0, stream>>>(sfg, rfg, sbg, rbg, dfgc, dbgc, far,
                                         f(cScale), f(cShift), SC,
                                         tmp, tmp+NRAYS*3, tmp+NRAYS*4, wfg, wbg);

  // hierarchical resampling
  k_sample_pdf<<<gRay, b256, 0, stream>>>(dfgc, wfg, dfgf, 101u);
  k_sample_pdf<<<gRay, b256, 0, stream>>>(dbgc, wbg, dbgf, 202u);

  // fine (sShift = log2(128) = 7)
  k_mlp<<<dim3(NRAYS*SF/64), b256, 0, stream>>>(f(iRo), dn, dfgf, 7, 0, mFf, w2t[2], sfg, rfg);
  k_mlp<<<dim3(NRAYS*SF/64), b256, 0, stream>>>(f(iRo), dn, dbgf, 7, 1, mFb, w2t[3], sbg, rbg);

  float* o = (float*)d_out;
  k_composite<<<gRay, b256, 0, stream>>>(sfg, rfg, sbg, rbg, dfgf, dbgf, far,
                                         f(fScale), f(fShift), SF,
                                         o, o+NRAYS*3, o+NRAYS*4, nullptr, nullptr);
}
